// MoEModel_41463614275837
// MI455X (gfx1250) — matrix operand order repaired and checked by reading, not compiled, not run on hardware
//
#include <hip/hip_runtime.h>
#include <hip/hip_bf16.h>
#include <math.h>

// ---------------------------------------------------------------------------
// MoE (top-1) for MI455X / gfx1250, wave32. conv2 = implicit GEMM on
// v_wmma_f32_16x16x32_bf16 with LDS-staged, fragment-native B slices.
// One workgroup (256 thr = 8 waves) per sample.
// ---------------------------------------------------------------------------

typedef __attribute__((ext_vector_type(16))) __bf16          v16bf;
typedef __attribute__((ext_vector_type(16))) unsigned short  v16us;
typedef __attribute__((ext_vector_type(8)))  unsigned short  v8us;
typedef __attribute__((ext_vector_type(8)))  float           v8f;

#define B_TOT   1024
#define C_IN    3
#define HW      32
#define C1      64
#define C2      128
#define GC      16
#define NCLS    100
#define K2      (C1 * 9)      /* 576 */
#define FC_IN   (C2 * 8 * 8)  /* 8192 */

#define H1P     18            /* padded h1 side */
#define H1PSZ   (C1 * H1P * H1P)   /* 20736 */
#define BS_STRIDE 40          /* ushorts per B-slice row: 16B aligned, bank-spread */

#define OUT_PROBS (B_TOT * NCLS)            /* 102400 */
#define OUT_AUX   (OUT_PROBS + B_TOT * 4)   /* 106496 */

static __device__ __forceinline__ unsigned short f32_to_bf16(float f) {
    unsigned int u = __float_as_uint(f);
    u += 0x7FFFu + ((u >> 16) & 1u);   // round-to-nearest-even
    return (unsigned short)(u >> 16);
}

static __device__ __forceinline__ v16bf as_bf16x16(v16us u) {
    union { v16us u; v16bf b; } c;
    c.u = u;
    return c.b;
}

// ---------------------------------------------------------------------------
// Gate kernel: conv(3->16,3x3,SAME) + relu + GAP + linear(16->4) + softmax,
// top-1 index/weight. One block per sample. Deterministic tree reduction.
// ---------------------------------------------------------------------------
__global__ __launch_bounds__(256)
void moe_gate_kernel(const float* __restrict__ x,
                     const float* __restrict__ gcw,
                     const float* __restrict__ gcb,
                     const float* __restrict__ gfw,
                     const float* __restrict__ gfb,
                     float* __restrict__ probs_out,
                     int*   __restrict__ idx_out,
                     float* __restrict__ wgt_out) {
    __shared__ float gx[C_IN * HW * HW];     // 12 KB
    __shared__ float gw[GC * 27];
    __shared__ float gb[GC];
    __shared__ float gred[GC * 256];         // 16 KB

    const int b   = blockIdx.x;
    const int tid = threadIdx.x;

    for (int i = tid; i < C_IN * HW * HW; i += 256) gx[i] = x[b * C_IN * HW * HW + i];
    for (int i = tid; i < GC * 27; i += 256)        gw[i] = gcw[i];
    if (tid < GC)                                   gb[tid] = gcb[tid];
    __syncthreads();

    float s[GC];
#pragma unroll
    for (int c = 0; c < GC; ++c) s[c] = 0.0f;

    for (int pi = 0; pi < 4; ++pi) {
        const int p  = tid + 256 * pi;
        const int y  = p >> 5;
        const int xx = p & 31;
        for (int c = 0; c < GC; ++c) {
            float acc = gb[c];
            for (int ci = 0; ci < C_IN; ++ci) {
#pragma unroll
                for (int ky = 0; ky < 3; ++ky) {
                    const int iy = y + ky - 1;
                    if ((unsigned)iy >= (unsigned)HW) continue;
#pragma unroll
                    for (int kx = 0; kx < 3; ++kx) {
                        const int ix = xx + kx - 1;
                        if ((unsigned)ix >= (unsigned)HW) continue;
                        acc += gx[ci * 1024 + iy * 32 + ix] * gw[c * 27 + ci * 9 + ky * 3 + kx];
                    }
                }
            }
            s[c] += fmaxf(acc, 0.0f);
        }
    }
#pragma unroll
    for (int c = 0; c < GC; ++c) gred[c * 256 + tid] = s[c];
    __syncthreads();
    for (int off = 128; off > 0; off >>= 1) {
        if (tid < off)
#pragma unroll
            for (int c = 0; c < GC; ++c)
                gred[c * 256 + tid] += gred[c * 256 + tid + off];
        __syncthreads();
    }

    if (tid == 0) {
        float logits[4];
#pragma unroll
        for (int e = 0; e < 4; ++e) {
            float l = gfb[e];
            for (int c = 0; c < GC; ++c)
                l += (gred[c * 256] * (1.0f / 1024.0f)) * gfw[e * GC + c];
            logits[e] = l;
        }
        float m = logits[0];
#pragma unroll
        for (int e = 1; e < 4; ++e) m = fmaxf(m, logits[e]);
        float pe[4], sum = 0.0f;
#pragma unroll
        for (int e = 0; e < 4; ++e) { pe[e] = expf(logits[e] - m); sum += pe[e]; }
        const float inv = 1.0f / sum;
        int   best_i = 0;
        float best_p = -1.0f;
#pragma unroll
        for (int e = 0; e < 4; ++e) {
            const float p = pe[e] * inv;
            probs_out[b * 4 + e] = p;
            if (p > best_p) { best_p = p; best_i = e; }
        }
        idx_out[b] = best_i;
        wgt_out[b] = best_p;
    }
}

// ---------------------------------------------------------------------------
// Expert kernel: one block per sample, runs ONLY the selected expert.
//  Stage A: conv1(3->64)+relu+pool2 in fp32 -> zero-padded h1 (bf16, LDS)
//  Stage B: conv2 implicit GEMM [128x576]x[576x256] on WMMA bf16:
//           per K-step, a cooperative pass builds the im2col slice in LDS
//           in fragment-native layout (B frags = 2x ds_load_b128, no
//           branches, no div chains); A frags stream from global (L2) f32
//           with in-register bf16 convert, reused across 16 N-tiles.
//  Stage C: bias+relu+pool2 epilogue -> h2, FC 8192->100 fp32, scale.
// ---------------------------------------------------------------------------
__global__ __launch_bounds__(256)
void moe_expert_kernel(const float* __restrict__ x,
                       const float* __restrict__ e_c1w,
                       const float* __restrict__ e_c1b,
                       const float* __restrict__ e_c2w,
                       const float* __restrict__ e_c2b,
                       const float* __restrict__ e_fw,
                       const float* __restrict__ e_fb,
                       const int*   __restrict__ sel_idx,
                       const float* __restrict__ sel_wgt,
                       float* __restrict__ out_final) {
    __shared__ float          lds_x[C_IN * HW * HW];      // 12 KB
    __shared__ float          lds_w1[C1 * 27];            // ~7 KB
    __shared__ float          lds_b1[C1];
    __shared__ float          lds_b2[C2];
    __shared__ unsigned short lds_h1p[H1PSZ];             // 41.5 KB padded bf16
    __shared__ unsigned short lds_koff[K2];               // 1.2 KB im2col offsets
    __shared__ unsigned short lds_bs[256 * BS_STRIDE];    // 20 KB B slice
    __shared__ float          lds_h2[C2 * 64];            // 32 KB
    __shared__ float          lds_red[256];

    const int b     = blockIdx.x;
    const int tid   = threadIdx.x;
    const int wave  = tid >> 5;
    const int lane  = tid & 31;
    const int lanex = lane & 15;
    const int lhi   = lane >> 4;      // 0: lanes 0-15, 1: lanes 16-31

    const int   e   = sel_idx[b];
    const float gwt = sel_wgt[b];

    // ---- stage load / init ----
    const float* xb = x + (size_t)b * C_IN * HW * HW;
    for (int i = tid; i < C_IN * HW * HW; i += 256) lds_x[i] = xb[i];
    const float* w1 = e_c1w + (size_t)e * C1 * 27;
    for (int i = tid; i < C1 * 27; i += 256) lds_w1[i] = w1[i];
    if (tid < C1) lds_b1[tid] = e_c1b[e * C1 + tid];
    if (tid < C2) lds_b2[tid] = e_c2b[e * C2 + tid];
    for (int i = tid; i < H1PSZ; i += 256) lds_h1p[i] = 0;          // zero pad ring
    for (int i = tid; i < C2 * 64; i += 256) lds_h2[i] = 0.0f;      // relu>=0 -> max-init 0
    for (int k = tid; k < K2; k += 256) {                            // im2col offset LUT
        const int ci = k / 9;
        const int rr = k - ci * 9;
        const int ky = rr / 3;
        const int kx = rr - ky * 3;
        lds_koff[k] = (unsigned short)(ci * (H1P * H1P) + ky * H1P + kx);
    }
    __syncthreads();

    // ---- stage A: conv1 + relu + maxpool2 -> h1p interior (bf16) ----
    for (int i = tid; i < C1 * 256; i += 256) {
        const int c  = i >> 8;
        const int p  = i & 255;
        const int py = p >> 4;
        const int px = p & 15;
        float mx = 0.0f;
#pragma unroll
        for (int dy = 0; dy < 2; ++dy) {
#pragma unroll
            for (int dx = 0; dx < 2; ++dx) {
                const int yy = py * 2 + dy;
                const int xx = px * 2 + dx;
                float s = lds_b1[c];
                for (int ci = 0; ci < C_IN; ++ci) {
#pragma unroll
                    for (int ky = 0; ky < 3; ++ky) {
                        const int iy = yy + ky - 1;
                        if ((unsigned)iy >= (unsigned)HW) continue;
#pragma unroll
                        for (int kx = 0; kx < 3; ++kx) {
                            const int ix = xx + kx - 1;
                            if ((unsigned)ix >= (unsigned)HW) continue;
                            s += lds_x[ci * 1024 + iy * 32 + ix] *
                                 lds_w1[c * 27 + ci * 9 + ky * 3 + kx];
                        }
                    }
                }
                mx = fmaxf(mx, fmaxf(s, 0.0f));
            }
        }
        lds_h1p[c * (H1P * H1P) + (py + 1) * H1P + (px + 1)] = f32_to_bf16(mx);
    }
    __syncthreads();

    // ---- stage B: implicit GEMM via WMMA bf16 ----
    // D[M=128, N=256], K=576 = 18 steps of 32. Wave w owns M-tile w.
    const int   mrow  = wave * 16 + lanex;
    const float* w2row = e_c2w + (size_t)e * C2 * K2 + (size_t)mrow * K2;

    // build-phase mapping: tid == output column n; y=n>>4, x=n&15
    const int rowbase = (tid >> 4) * H1P + (tid & 15);   // into padded h1 (koff adds rest)

    v8f acc[16];
#pragma unroll
    for (int nt = 0; nt < 16; ++nt)
#pragma unroll
        for (int q = 0; q < 8; ++q) acc[nt][q] = 0.0f;

    for (int ks = 0; ks < K2 / 32; ++ks) {
        const int kbase = ks * 32;

        // --- cooperative B-slice build: Bs[n][kk], kk = K-kbase in [0,32) ---
#pragma unroll
        for (int g = 0; g < 2; ++g) {
            v8us t0, t1;
#pragma unroll
            for (int i2 = 0; i2 < 8; ++i2)
                t0[i2] = lds_h1p[lds_koff[kbase + g * 16 + i2] + rowbase];
#pragma unroll
            for (int i2 = 0; i2 < 8; ++i2)
                t1[i2] = lds_h1p[lds_koff[kbase + g * 16 + 8 + i2] + rowbase];
            *(v8us*)&lds_bs[tid * BS_STRIDE + g * 16]     = t0;
            *(v8us*)&lds_bs[tid * BS_STRIDE + g * 16 + 8] = t1;
        }
        __syncthreads();

        // --- A fragment from global f32 (L2) -> bf16, reused over 16 N-tiles ---
        // elem i<8  : K = kbase + 8*lhi + i        (VGPRs 0-3)
        // elem i>=8 : K = kbase + 16 + 8*lhi + i-8 (VGPRs 4-7)
        const float4 a0 = *(const float4*)(w2row + kbase + lhi * 8);
        const float4 a1 = *(const float4*)(w2row + kbase + lhi * 8 + 4);
        const float4 a2 = *(const float4*)(w2row + kbase + 16 + lhi * 8);
        const float4 a3 = *(const float4*)(w2row + kbase + 16 + lhi * 8 + 4);
        v16us au;
        au[0]  = f32_to_bf16(a0.x); au[1]  = f32_to_bf16(a0.y);
        au[2]  = f32_to_bf16(a0.z); au[3]  = f32_to_bf16(a0.w);
        au[4]  = f32_to_bf16(a1.x); au[5]  = f32_to_bf16(a1.y);
        au[6]  = f32_to_bf16(a1.z); au[7]  = f32_to_bf16(a1.w);
        au[8]  = f32_to_bf16(a2.x); au[9]  = f32_to_bf16(a2.y);
        au[10] = f32_to_bf16(a2.z); au[11] = f32_to_bf16(a2.w);
        au[12] = f32_to_bf16(a3.x); au[13] = f32_to_bf16(a3.y);
        au[14] = f32_to_bf16(a3.z); au[15] = f32_to_bf16(a3.w);
        const v16bf abf = as_bf16x16(au);

        // --- consume: B frag = 2x ds_load_b128, elem i<8 K = kbase+8*lhi+i, i>=8 K = kbase+16+8*lhi+i-8
#pragma unroll
        for (int nt = 0; nt < 16; ++nt) {
            const int n   = nt * 16 + lanex;
            const int off = n * BS_STRIDE + 8 * lhi;
            union { v16us v; struct { v8us lo, hi; } s; } bb;
            bb.s.lo = *(const v8us*)&lds_bs[off];
            bb.s.hi = *(const v8us*)&lds_bs[off + 16];
            acc[nt] = __builtin_amdgcn_wmma_f32_16x16x32_bf16(
                false, abf, false, as_bf16x16(bb.v),
                (short)0, acc[nt], false, false);
        }
        __syncthreads();   // protect Bs before next build
    }

    // ---- epilogue: bias + relu + maxpool2 into h2[128][8][8] ----
    // C/D layout: VGPR r -> M=r (lanes 0-15) / M=r+8 (lanes 16-31), N=lane%16
#pragma unroll
    for (int nt = 0; nt < 16; ++nt) {
        const int py = nt >> 1;
        const int px = lanex >> 1;
#pragma unroll
        for (int r = 0; r < 8; ++r) {
            const int cout = wave * 16 + r + lhi * 8;
            float v = acc[nt][r] + lds_b2[cout];
            v = fmaxf(v, 0.0f);
            atomicMax((int*)&lds_h2[cout * 64 + py * 8 + px], __float_as_int(v));
        }
    }
    __syncthreads();

    // ---- stage C: FC 8192 -> 100, scale by router weight ----
    const int outc = tid >> 1;
    const int half = tid & 1;
    float partial = 0.0f;
    if (outc < NCLS) {
        const float4* fw4 = reinterpret_cast<const float4*>(
            e_fw + ((size_t)e * NCLS + outc) * FC_IN + half * 4096);
        const float4* h4  = reinterpret_cast<const float4*>(lds_h2 + half * 4096);
        for (int i = 0; i < 1024; ++i) {
            const float4 w = fw4[i];
            const float4 h = h4[i];
            partial += w.x * h.x + w.y * h.y + w.z * h.z + w.w * h.w;
        }
    }
    lds_red[tid] = partial;
    __syncthreads();
    if (half == 0 && outc < NCLS) {
        const float tot = lds_red[tid] + lds_red[tid + 1] + e_fb[e * NCLS + outc];
        out_final[(size_t)b * NCLS + outc] = gwt * tot;
    }
}

// ---------------------------------------------------------------------------
// Aux-loss kernel: deterministic fixed-partition reduction over router probs.
// ---------------------------------------------------------------------------
__global__ __launch_bounds__(256)
void moe_aux_kernel(const float* __restrict__ probs, float* __restrict__ aux_out) {
    __shared__ float red[4 * 256];
    const int tid = threadIdx.x;
    float a[4] = {0.f, 0.f, 0.f, 0.f};
    for (int bb = tid; bb < B_TOT; bb += 256)
#pragma unroll
        for (int e = 0; e < 4; ++e) a[e] += probs[bb * 4 + e];
#pragma unroll
    for (int e = 0; e < 4; ++e) red[e * 256 + tid] = a[e];
    __syncthreads();
    for (int off = 128; off > 0; off >>= 1) {
        if (tid < off)
#pragma unroll
            for (int e = 0; e < 4; ++e)
                red[e * 256 + tid] += red[e * 256 + tid + off];
        __syncthreads();
    }
    if (tid == 0) {
        float s = 0.0f;
#pragma unroll
        for (int e = 0; e < 4; ++e) {
            const float d = red[e * 256] * (1.0f / 1024.0f) - 0.25f;
            s += d * d;
        }
        aux_out[0] = s * 0.25f;
    }
}

// ---------------------------------------------------------------------------
extern "C" void kernel_launch(void* const* d_in, const int* in_sizes, int n_in,
                              void* d_out, int out_size, void* d_ws, size_t ws_size,
                              hipStream_t stream) {
    const float* x       = (const float*)d_in[0];
    const float* gate_cw = (const float*)d_in[1];
    const float* gate_cb = (const float*)d_in[2];
    const float* gate_fw = (const float*)d_in[3];
    const float* gate_fb = (const float*)d_in[4];
    const float* e_c1w   = (const float*)d_in[5];
    const float* e_c1b   = (const float*)d_in[6];
    const float* e_c2w   = (const float*)d_in[7];
    const float* e_c2b   = (const float*)d_in[8];
    const float* e_fw    = (const float*)d_in[9];
    const float* e_fb    = (const float*)d_in[10];

    float* out       = (float*)d_out;
    float* probs_out = out + OUT_PROBS;
    float* aux_out   = out + OUT_AUX;

    int*   sel_idx = (int*)d_ws;
    float* sel_wgt = (float*)((char*)d_ws + 4096);

    moe_gate_kernel<<<B_TOT, 256, 0, stream>>>(x, gate_cw, gate_cb, gate_fw, gate_fb,
                                               probs_out, sel_idx, sel_wgt);
    moe_expert_kernel<<<B_TOT, 256, 0, stream>>>(x, e_c1w, e_c1b, e_c2w, e_c2b,
                                                 e_fw, e_fb, sel_idx, sel_wgt, out);
    moe_aux_kernel<<<1, 256, 0, stream>>>(probs_out, aux_out);
}
